// TSPModel_18811956756958
// MI455X (gfx1250) — matrix operand order repaired and checked by reading, not compiled, not run on hardware
//
#include <hip/hip_runtime.h>

#define B_   64
#define P_   100
#define N_   1000
#define D_   128
#define H_   8
#define DK_  16
#define HID_ 256
#define PPAD 112       // P padded to 7*16
#define PT_  7
#define NT16 63        // ceil(1000/16)
#define NPAD 1008      // 63*16
#define VTW  1024      // padded width of transposed V
#define NEGV (-1e9f)

typedef __attribute__((ext_vector_type(16))) __bf16 v16bf;
typedef __attribute__((ext_vector_type(8)))  __bf16 v8bf;
typedef __attribute__((ext_vector_type(8)))  float  v8f;

static __device__ __forceinline__ v8f wmma_bf16(v16bf a, v16bf b, v8f c) {
  // D = A(16x32 bf16) x B(32x16 bf16) + C(16x16 f32)
  return __builtin_amdgcn_wmma_f32_16x16x32_bf16(false, a, false, b, (short)0, c,
                                                 false, false);
}

// ---- CDNA5 async global->LDS copy (ASYNCcnt path) ------------------------
static __device__ __forceinline__ unsigned lds_addr32(const void* p) {
  // generic shared pointer: LDS byte address is addr[31:0]
  return (unsigned)(unsigned long long)p;
}
static __device__ __forceinline__ void async_load_b128(unsigned lds_addr,
                                                       const void* gptr) {
  asm volatile("global_load_async_to_lds_b128 %0, %1, off"
               :
               : "v"(lds_addr), "v"((unsigned long long)gptr)
               : "memory");
}
static __device__ __forceinline__ void wait_asynccnt0() {
  asm volatile("s_wait_asynccnt 0" ::: "memory");
}

// ---- fragment loaders (wave32; contiguous K order, k = 16*(lane/16) + j) --------------
// A 16x32 bf16: lane<16 -> M=lane, elems j = K j
//               lane>=16 -> M=lane-16, elems j = K 16+j
static __device__ __forceinline__ v16bf a_frag_f32(const float* __restrict__ src,
                                                   int rbase, int rlimit, int ld,
                                                   int cbase) {
  int lane = threadIdx.x & 31;
  int m = lane & 15;
  int kb = (lane < 16) ? 0 : 16;
  int row = rbase + m;
  v16bf a;
  if (row < rlimit) {
    const float* p = src + (size_t)row * ld + cbase + kb;
#pragma unroll
    for (int e = 0; e < 8; ++e) { a[e] = (__bf16)p[e]; a[e + 8] = (__bf16)p[e + 8]; }
  } else {
#pragma unroll
    for (int e = 0; e < 16; ++e) a[e] = (__bf16)0.0f;
  }
  return a;
}

static __device__ __forceinline__ v16bf a_frag_bf16(const __bf16* __restrict__ src,
                                                    int rbase, int ld, int cbase) {
  int lane = threadIdx.x & 31;
  int m = lane & 15;
  int kb = (lane < 16) ? 0 : 16;
  const __bf16* p = src + (size_t)(rbase + m) * ld + cbase + kb;
  v16bf a;
#pragma unroll
  for (int e = 0; e < 8; ++e) { a[e] = p[e]; a[e + 8] = p[e + 8]; }
  return a;
}

// B 32x16 bf16: lane<16 -> N=lane,   elems j = K j
//               lane>=16 -> N=lane-16, elems j = K 16+j
// Source is row-major [out=N][in=K]: per-lane 16 contiguous bf16.
static __device__ __forceinline__ v16bf b_frag_bf16(const __bf16* __restrict__ src,
                                                    int nbase, int ld, int cbase) {
  int lane = threadIdx.x & 31;
  int n = lane & 15;
  int kb = (lane < 16) ? 0 : 16;
  const __bf16* p = src + (size_t)(nbase + n) * ld + cbase + kb;
  v16bf b;
#pragma unroll
  for (int j = 0; j < 16; ++j) b[j] = p[j];
  return b;
}

static __device__ __forceinline__ v16bf b_frag_f32_guard(const float* __restrict__ src,
                                                         int nbase, int nlimit, int ld,
                                                         int cbase) {
  int lane = threadIdx.x & 31;
  int n = lane & 15;
  int kb = (lane < 16) ? 0 : 16;
  int row = nbase + n;
  v16bf b;
  if (row < nlimit) {
    const float* p = src + (size_t)row * ld + cbase + kb;
#pragma unroll
    for (int j = 0; j < 16; ++j) b[j] = (__bf16)p[j];
  } else {
#pragma unroll
    for (int j = 0; j < 16; ++j) b[j] = (__bf16)0.0f;
  }
  return b;
}

// ---- K1: hypernetwork MLP -> mid[10] -------------------------------------
__global__ __launch_bounds__(256) void hyper_kernel(
    const float* __restrict__ pref,
    const float* __restrict__ fc1_w, const float* __restrict__ fc1_b,
    const float* __restrict__ fc2_w, const float* __restrict__ fc2_b,
    const float* __restrict__ fc3_w, const float* __restrict__ fc3_b,
    float* __restrict__ mid) {
  __shared__ float h1[HID_];
  __shared__ float h2[HID_];
  int t = threadIdx.x;
  h1[t] = pref[0] * fc1_w[t * 2 + 0] + pref[1] * fc1_w[t * 2 + 1] + fc1_b[t];
  __syncthreads();
  float s = fc2_b[t];
  for (int j = 0; j < HID_; ++j) s += fc2_w[t * HID_ + j] * h1[j];
  h2[t] = s;
  __syncthreads();
  if (t < 10) {
    float s2 = fc3_b[t];
    for (int j = 0; j < HID_; ++j) s2 += fc3_w[t * HID_ + j] * h2[j];
    mid[t] = s2;
  }
}

// ---- K2: expand 5x (128x128) weight matrices to bf16 ---------------------
__global__ __launch_bounds__(256) void weight_kernel(
    const float* __restrict__ mid,
    const float* __restrict__ w0, const float* __restrict__ w1,
    const float* __restrict__ w2, const float* __restrict__ w3,
    const float* __restrict__ w4, __bf16* __restrict__ Wbf) {
  int c = blockIdx.y;
  int j = blockIdx.x * 256 + threadIdx.x;   // 0..16383
  const float* src = (c == 0) ? w0 : (c == 1) ? w1 : (c == 2) ? w2 : (c == 3) ? w3 : w4;
  float v = mid[2 * c] * src[j * 2 + 0] + mid[2 * c + 1] * src[j * 2 + 1];
  Wbf[c * (D_ * D_) + j] = (__bf16)v;
}

// ---- K3a: q = q1 @ Wqf^T + last @ Wql^T -> qbf [B][H][112][16] ----------
__global__ __launch_bounds__(256) void q_kernel(
    const float* __restrict__ q1, const float* __restrict__ qlast,
    const __bf16* __restrict__ Wbf, __bf16* __restrict__ qbf) {
  int b = blockIdx.y, pbase = blockIdx.x * 16;
  int wv = threadIdx.x >> 5, lane = threadIdx.x & 31;
  int m = lane & 15, mb = (lane < 16) ? 0 : 8;
  const float* q1b = q1 + (size_t)b * P_ * D_;
  const float* qlb = qlast + (size_t)b * P_ * D_;
  v8f c = {};
#pragma unroll
  for (int kt = 0; kt < 4; ++kt) {
    int cb = kt * 32;
    v16bf a1 = a_frag_f32(q1b, pbase, P_, D_, cb);
    v16bf a2 = a_frag_f32(qlb, pbase, P_, D_, cb);
    v16bf bq1 = b_frag_bf16(Wbf + 0 * (D_ * D_), wv * 16, D_, cb);
    v16bf bq2 = b_frag_bf16(Wbf + 1 * (D_ * D_), wv * 16, D_, cb);
    c = wmma_bf16(a1, bq1, c);
    c = wmma_bf16(a2, bq2, c);
  }
  __bf16* dst = qbf + ((size_t)b * H_ + wv) * PPAD * DK_;
#pragma unroll
  for (int r = 0; r < 8; ++r) {
    int p = pbase + mb + r;                 // < 112 always
    dst[p * DK_ + m] = (__bf16)c[r];
  }
}

// ---- K3b: k,v projections; v stored transposed [B][H][16][1024] ---------
__global__ __launch_bounds__(256) void kv_kernel(
    const float* __restrict__ nodes, const __bf16* __restrict__ Wbf,
    __bf16* __restrict__ kbf, __bf16* __restrict__ vT) {
  int b = blockIdx.y, nbase = blockIdx.x * 16;
  int wv = threadIdx.x >> 5, lane = threadIdx.x & 31;
  int m = lane & 15, mb = (lane < 16) ? 0 : 8;
  const float* nb = nodes + (size_t)b * N_ * D_;
  v8f ck = {}, cv = {};
#pragma unroll
  for (int kt = 0; kt < 4; ++kt) {
    int cb = kt * 32;
    v16bf a = a_frag_f32(nb, nbase, N_, D_, cb);
    v16bf bk = b_frag_bf16(Wbf + 2 * (D_ * D_), wv * 16, D_, cb);
    v16bf bv = b_frag_bf16(Wbf + 3 * (D_ * D_), wv * 16, D_, cb);
    ck = wmma_bf16(a, bk, ck);
    cv = wmma_bf16(a, bv, cv);
  }
  size_t bh = (size_t)b * H_ + wv;
#pragma unroll
  for (int r = 0; r < 8; ++r) {
    int n = nbase + mb + r;                 // < 1008 always (pad rows are 0)
    kbf[(bh * NPAD + n) * DK_ + m] = (__bf16)ck[r];
  }
  v8bf pack;
#pragma unroll
  for (int r = 0; r < 8; ++r) pack[r] = (__bf16)cv[r];
  *reinterpret_cast<v8bf*>(vT + (bh * 16 + m) * VTW + nbase + mb) = pack;
}

// ---- K4: fused masked attention per (ptile, head, batch) -----------------
__global__ __launch_bounds__(256) void attn_kernel(
    const __bf16* __restrict__ qbf, const __bf16* __restrict__ kbf,
    const __bf16* __restrict__ vT, const float* __restrict__ mask,
    __bf16* __restrict__ oc) {
  extern __shared__ char smem[];
  float*  score   = (float*)smem;                               // 16*1024 f32
  __bf16* probs   = (__bf16*)(smem + 16 * 1024 * 4);            // 16*1024 bf16
  float*  rowinv  = (float*)(smem + 16 * 1024 * 4 + 16 * 1024 * 2); // 16
  float*  outtile = rowinv + 16;                                // 16*16 f32

  int pbase = blockIdx.x * 16, h = blockIdx.y, b = blockIdx.z;
  int tid = threadIdx.x, wv = tid >> 5, lane = tid & 31;
  int m = lane & 15, mb = (lane < 16) ? 0 : 8;
  int kb8 = (lane < 16) ? 0 : 8, kb16 = (lane < 16) ? 0 : 16;
  size_t bh = (size_t)b * H_ + h;

  // Stage the mask stripe [rowsValid x 1000] f32 into the score LDS buffer
  // with the async global->LDS engine (overlapped with q fragment load).
  {
    int rowsValid = P_ - pbase; if (rowsValid > 16) rowsValid = 16;
    int total = rowsValid * 250;                 // 250 float4 per row
    for (int idx = tid; idx < total; idx += 256) {
      int row = idx / 250, c4 = idx % 250;
      const float* g = mask + ((size_t)b * P_ + pbase + row) * N_ + c4 * 4;
      async_load_b128(lds_addr32(score + row * 1024 + c4 * 4), g);
    }
  }

  // q A-fragment, K padded 16 -> 32 with zeros
  v16bf aq;
  {
    const __bf16* qp = qbf + (bh * PPAD + pbase + m) * DK_;
#pragma unroll
    for (int e = 0; e < 8; ++e) { aq[e] = (lane < 16) ? qp[e] : (__bf16)0.0f; aq[e + 8] = (lane < 16) ? qp[e + 8] : (__bf16)0.0f; }
  }

  wait_asynccnt0();
  __syncthreads();

  // Phase 1: score = (q kT)/4 + mask (in-place over the staged mask)
  for (int nt = wv; nt < NT16; nt += 8) {
    if (nt + 8 < NT16)
      __builtin_prefetch(kbf + (bh * NPAD + (nt + 8) * 16 + m) * DK_, 0, 3);
    v16bf bk;
    if (lane < 16) {
      const __bf16* kp = kbf + (bh * NPAD + nt * 16 + m) * DK_;
#pragma unroll
      for (int j = 0; j < 16; ++j) bk[j] = kp[j];
    } else {
#pragma unroll
      for (int j = 0; j < 16; ++j) bk[j] = (__bf16)0.0f;   // K 16..31 pad
    }
    v8f c = {};
    c = wmma_bf16(aq, bk, c);
#pragma unroll
    for (int r = 0; r < 8; ++r) {
      int mm = mb + r, p = pbase + mm, n = nt * 16 + m;
      float msk = (n < N_) ? ((p < P_) ? score[mm * 1024 + n] : 0.0f) : NEGV;
      score[mm * 1024 + n] = c[r] * 0.25f + msk;
    }
  }
  __syncthreads();

  // Phase 2: per-row softmax (unnormalized e; 1/sum kept separately)
  for (int rr = 2 * wv; rr < 2 * wv + 2; ++rr) {
    int p = pbase + rr;
    float* srow = score + rr * 1024;
    __bf16* prow = probs + rr * 1024;
    if (p < P_) {
      float mx = -3.4e38f;
      for (int n = lane; n < N_; n += 32) mx = fmaxf(mx, srow[n]);
#pragma unroll
      for (int o = 16; o >= 1; o >>= 1) mx = fmaxf(mx, __shfl_xor(mx, o, 32));
      float sum = 0.0f;
      for (int n = lane; n < N_; n += 32) {
        float e = __expf(srow[n] - mx);
        prow[n] = (__bf16)e;
        sum += e;
      }
#pragma unroll
      for (int o = 16; o >= 1; o >>= 1) sum += __shfl_xor(sum, o, 32);
      for (int n = N_ + lane; n < 1024; n += 32) prow[n] = (__bf16)0.0f;
      if (lane == 0) rowinv[rr] = 1.0f / sum;
    } else {
      for (int n = lane; n < 1024; n += 32) prow[n] = (__bf16)0.0f;
      if (lane == 0) rowinv[rr] = 0.0f;
    }
  }
  if (tid < 256) outtile[tid] = 0.0f;
  __syncthreads();

  // Phase 3: out = probs x V (chunks of 32 over N), cross-wave ds_add reduce
  for (int ch = wv; ch < 32; ch += 8) {
    v16bf ap;
    {
      const __bf16* pp = probs + m * 1024 + ch * 32 + kb16;
#pragma unroll
      for (int e = 0; e < 8; ++e) { ap[e] = pp[e]; ap[e + 8] = pp[e + 8]; }
    }
    v16bf bv;
    int n0 = ch * 32 + kb16;
    if (n0 < NPAD) {
      const __bf16* vp = vT + (bh * 16 + m) * VTW + n0;
#pragma unroll
      for (int j = 0; j < 16; ++j) bv[j] = vp[j];
    } else {
#pragma unroll
      for (int j = 0; j < 16; ++j) bv[j] = (__bf16)0.0f;
    }
    v8f c = {};
    c = wmma_bf16(ap, bv, c);
#pragma unroll
    for (int r = 0; r < 8; ++r) atomicAdd(&outtile[(mb + r) * 16 + m], c[r]);
  }
  __syncthreads();

  if (tid < 256) {
    int mm = tid >> 4, d = tid & 15;
    float v = outtile[tid] * rowinv[mm];
    oc[((size_t)b * PPAD + pbase + mm) * D_ + h * DK_ + d] = (__bf16)v;
  }
}

// ---- K5: mh = oc @ Wmhc^T; logits = 10*tanh(mh nodes^T / sqrtD) + mask;
//          softmax over N -> d_out -------------------------------------
__global__ __launch_bounds__(256) void final_kernel(
    const __bf16* __restrict__ oc, const __bf16* __restrict__ Wbf,
    const float* __restrict__ nodes, const float* __restrict__ mask,
    float* __restrict__ out) {
  extern __shared__ char smem[];
  __bf16* mh = (__bf16*)smem;                 // 16*128 bf16
  float* logits = (float*)(smem + 4096);      // 16*1024 f32

  int pbase = blockIdx.x * 16, b = blockIdx.y;
  int tid = threadIdx.x, wv = tid >> 5, lane = tid & 31;
  int m = lane & 15, mb = (lane < 16) ? 0 : 8;
  const __bf16* Wmhc = Wbf + 4 * (D_ * D_);
  const float* nb = nodes + (size_t)b * N_ * D_;

  // Stage mask stripe into logits LDS buffer, overlapped with Phase A WMMAs.
  {
    int rowsValid = P_ - pbase; if (rowsValid > 16) rowsValid = 16;
    int total = rowsValid * 250;
    for (int idx = tid; idx < total; idx += 256) {
      int row = idx / 250, c4 = idx % 250;
      const float* g = mask + ((size_t)b * P_ + pbase + row) * N_ + c4 * 4;
      async_load_b128(lds_addr32(logits + row * 1024 + c4 * 4), g);
    }
  }

  // Phase A: mh tile [16 x 128], wave wv owns output cols [16wv,16wv+16)
  v8f c = {};
#pragma unroll
  for (int kt = 0; kt < 4; ++kt) {
    v16bf a = a_frag_bf16(oc + (size_t)b * PPAD * D_, pbase, D_, kt * 32);
    v16bf bb = b_frag_bf16(Wmhc, wv * 16, D_, kt * 32);
    c = wmma_bf16(a, bb, c);
  }
#pragma unroll
  for (int r = 0; r < 8; ++r) mh[(mb + r) * D_ + wv * 16 + m] = (__bf16)c[r];
  wait_asynccnt0();
  __syncthreads();

  // Phase B: score2 = mh @ nodes^T ; clipped tanh + mask (in-place) -> logits
  for (int nt = wv; nt < NT16; nt += 8) {
    if (nt + 8 < NT16)
      __builtin_prefetch(nb + (size_t)((nt + 8) * 16 + m) * D_, 0, 3);
    v8f c2 = {};
#pragma unroll
    for (int kt = 0; kt < 4; ++kt) {
      v16bf a = a_frag_bf16(mh, 0, D_, kt * 32);
      v16bf bb = b_frag_f32_guard(nb, nt * 16, N_, D_, kt * 32);
      c2 = wmma_bf16(a, bb, c2);
    }
#pragma unroll
    for (int r = 0; r < 8; ++r) {
      int mm = mb + r, p = pbase + mm, n = nt * 16 + m;
      float lg = 10.0f * tanhf(c2[r] * 0.08838834764831845f);  // 1/sqrt(128)
      float msk = (n < N_) ? ((p < P_) ? logits[mm * 1024 + n] : 0.0f) : NEGV;
      logits[mm * 1024 + n] = lg + msk;
    }
  }
  __syncthreads();

  // Phase C: row softmax -> global output
  for (int rr = 2 * wv; rr < 2 * wv + 2; ++rr) {
    int p = pbase + rr;
    if (p >= P_) continue;
    float* lrow = logits + rr * 1024;
    float mx = -3.4e38f;
    for (int n = lane; n < N_; n += 32) mx = fmaxf(mx, lrow[n]);
#pragma unroll
    for (int o = 16; o >= 1; o >>= 1) mx = fmaxf(mx, __shfl_xor(mx, o, 32));
    float sum = 0.0f;
    for (int n = lane; n < N_; n += 32) {
      float e = __expf(lrow[n] - mx);
      lrow[n] = e;
      sum += e;
    }
#pragma unroll
    for (int o = 16; o >= 1; o >>= 1) sum += __shfl_xor(sum, o, 32);
    float inv = 1.0f / sum;
    for (int n = lane; n < N_; n += 32)
      out[((size_t)b * P_ + p) * N_ + n] = lrow[n] * inv;
  }
}

// --------------------------------------------------------------------------
extern "C" void kernel_launch(void* const* d_in, const int* in_sizes, int n_in,
                              void* d_out, int out_size, void* d_ws, size_t ws_size,
                              hipStream_t stream) {
  (void)in_sizes; (void)n_in; (void)out_size; (void)ws_size;
  const float* pref   = (const float*)d_in[0];
  const float* nodes  = (const float*)d_in[1];
  const float* q1     = (const float*)d_in[2];
  const float* qlast  = (const float*)d_in[3];
  const float* mask   = (const float*)d_in[4];
  const float* fc1_w  = (const float*)d_in[5];
  const float* fc1_b  = (const float*)d_in[6];
  const float* fc2_w  = (const float*)d_in[7];
  const float* fc2_b  = (const float*)d_in[8];
  const float* fc3_w  = (const float*)d_in[9];
  const float* fc3_b  = (const float*)d_in[10];
  const float* wqf    = (const float*)d_in[11];
  const float* wql    = (const float*)d_in[12];
  const float* wk     = (const float*)d_in[13];
  const float* wvw    = (const float*)d_in[14];
  const float* wmhc   = (const float*)d_in[15];
  float* out = (float*)d_out;

  // workspace layout (256B aligned)
  char* ws = (char*)d_ws;
  size_t off = 0;
  auto take = [&](size_t bytes) {
    size_t o = off;
    off = (off + bytes + 255) & ~(size_t)255;
    return o;
  };
  float*  mid = (float*)(ws + take(16 * sizeof(float)));
  __bf16* Wbf = (__bf16*)(ws + take((size_t)5 * D_ * D_ * 2));
  __bf16* qbf = (__bf16*)(ws + take((size_t)B_ * H_ * PPAD * DK_ * 2));
  __bf16* kbf = (__bf16*)(ws + take((size_t)B_ * H_ * NPAD * DK_ * 2));
  __bf16* vT  = (__bf16*)(ws + take((size_t)B_ * H_ * 16 * VTW * 2));
  __bf16* oc  = (__bf16*)(ws + take((size_t)B_ * PPAD * D_ * 2));

  const int ATTN_SMEM = 16 * 1024 * 4 + 16 * 1024 * 2 + 16 * 4 + 256 * 4; // 99392
  const int FIN_SMEM  = 4096 + 16 * 1024 * 4;                             // 69632
  hipFuncSetAttribute((const void*)attn_kernel,
                      hipFuncAttributeMaxDynamicSharedMemorySize, ATTN_SMEM);
  hipFuncSetAttribute((const void*)final_kernel,
                      hipFuncAttributeMaxDynamicSharedMemorySize, FIN_SMEM);

  hyper_kernel<<<1, 256, 0, stream>>>(pref, fc1_w, fc1_b, fc2_w, fc2_b, fc3_w,
                                      fc3_b, mid);
  weight_kernel<<<dim3(64, 5), 256, 0, stream>>>(mid, wqf, wql, wk, wvw, wmhc, Wbf);
  q_kernel<<<dim3(PT_, B_), 256, 0, stream>>>(q1, qlast, Wbf, qbf);
  kv_kernel<<<dim3(NT16, B_), 256, 0, stream>>>(nodes, Wbf, kbf, vT);
  attn_kernel<<<dim3(PT_, H_, B_), 256, ATTN_SMEM, stream>>>(qbf, kbf, vT, mask, oc);
  final_kernel<<<dim3(PT_, B_), 256, FIN_SMEM, stream>>>(oc, Wbf, nodes, mask, out);
}
